// RecurrentActorCritic_50611894616081
// MI455X (gfx1250) — compile-verified, matrix operand order checked by reading, not run on hardware
//
#include <hip/hip_runtime.h>

// ---------------------------------------------------------------------------
// RecurrentActorCritic on gfx1250: fused WMMA-f16 LSTM scan.
// B=256, T=128, D=256, H=512, A=8.
// ---------------------------------------------------------------------------

typedef _Float16 f16t;
typedef __attribute__((ext_vector_type(16))) _Float16 v16h;
typedef __attribute__((ext_vector_type(8)))  _Float16 v8h;
typedef __attribute__((ext_vector_type(8)))  float    v8f;

union V16U { v16h v; v8h h[2]; };

constexpr int B_ = 256;
constexpr int T_ = 128;
constexpr int D_ = 256;
constexpr int H_ = 512;
constexpr int A_ = 8;

__device__ __forceinline__ float sigmoidf_(float x) {
    return 1.0f / (1.0f + __expf(-x));
}

// ---------------------------------------------------------------------------
// f32 -> f16 conversion
// ---------------------------------------------------------------------------
__global__ void cvt_f32_f16_kernel(const float* __restrict__ src,
                                   f16t* __restrict__ dst, int n) {
    int i = blockIdx.x * blockDim.x + threadIdx.x;
    if (i < n) dst[i] = (f16t)src[i];
}

// ---------------------------------------------------------------------------
// Pack [Wh (H x 4H); Wx (KX x 4H)] into transposed f16 [4H x (H+KX)]:
//   dst[n*Ktot + k] = k<H ? Wh[k,n] : Wx[k-H,n]
// ---------------------------------------------------------------------------
__global__ void pack_weights_kernel(const float* __restrict__ Wh,
                                    const float* __restrict__ Wx,
                                    f16t* __restrict__ dst,
                                    int K1, int Kx) {
    const int Ktot = K1 + Kx;
    int idx = blockIdx.x * blockDim.x + threadIdx.x;
    if (idx >= 4 * H_ * Ktot) return;
    int n = idx / Ktot;
    int k = idx - n * Ktot;
    float v = (k < K1) ? Wh[(size_t)k * (4 * H_) + n]
                       : Wx[(size_t)(k - K1) * (4 * H_) + n];
    dst[idx] = (f16t)v;
}

// ---------------------------------------------------------------------------
// One LSTM time step for TWO independent LSTMs (blockIdx.y selects
// critic / actor), fully fused:
//   z = [h_prev | seq[:,t]] @ WcT' + bias     (WMMA f16 -> f32)
//   i,f,g,o gate math + cell update in epilogue.
// Each wave computes one 16(M) x 16(N over H) tile and accumulates the 4
// gate tiles (N, H+N, 2H+N, 3H+N) so the nonlinearity fuses in-register.
// K loop is split into the recurrent segment (K=H) and input segment (K=KX)
// with compile-time bounds so addresses fold to base+imm and loads pipeline.
// grid = dim3(64, 2), block = 256 threads (8 wave32).
// ---------------------------------------------------------------------------
template <int KX, bool HAS_SEQ_OUT>
__global__ __launch_bounds__(256)
void lstm_step_kernel(const f16t* __restrict__ hprev0, const f16t* __restrict__ hprev1,
                      const f16t* __restrict__ seq0,   const f16t* __restrict__ seq1,
                      const f16t* __restrict__ WcT0,   const f16t* __restrict__ WcT1,
                      const float* __restrict__ bias0, const float* __restrict__ bias1,
                      float* __restrict__ cst0,  float* __restrict__ cst1,
                      float* __restrict__ hst0,  float* __restrict__ hst1,
                      f16t* __restrict__ hout0,  f16t* __restrict__ hout1,
                      f16t* __restrict__ oseq0,  f16t* __restrict__ oseq1,
                      int t) {
    constexpr int KTOT = H_ + KX;

    const bool second = (blockIdx.y != 0);
    const f16t* hprev = second ? hprev1 : hprev0;
    const f16t* seq   = second ? seq1   : seq0;
    const f16t* WcT   = second ? WcT1   : WcT0;
    const float* bias = second ? bias1  : bias0;
    float* c_state    = second ? cst1   : cst0;
    float* h_state    = second ? hst1   : hst0;
    f16t* hout        = second ? hout1  : hout0;
    f16t* outseq      = second ? oseq1  : oseq0;

    const int lane = threadIdx.x & 31;
    const int wv   = threadIdx.x >> 5;
    const int tile = blockIdx.x * 8 + wv;
    const int m0   = (tile >> 5) << 4;      // 16 tiles over B
    const int n0   = (tile & 31) << 4;      // 32 tiles over H
    const int l15  = lane & 15;
    const int khi  = (lane < 16) ? 0 : 8;   // K sub-chunk per lane half

    // Per-lane base pointers (constant stride walk; no per-iter selects).
    const f16t* ap  = hprev + (size_t)(m0 + l15) * H_ + khi;
    const f16t* xp  = seq + ((size_t)(m0 + l15) * T_ + t) * KX + khi;
    const f16t* bp0 = WcT + (size_t)(0 * H_ + n0 + l15) * KTOT + khi;
    const f16t* bp1 = WcT + (size_t)(1 * H_ + n0 + l15) * KTOT + khi;
    const f16t* bp2 = WcT + (size_t)(2 * H_ + n0 + l15) * KTOT + khi;
    const f16t* bp3 = WcT + (size_t)(3 * H_ + n0 + l15) * KTOT + khi;

    v8f acc[4];
#pragma unroll
    for (int g = 0; g < 4; ++g) {
        float bg = bias[g * H_ + n0 + l15];
        acc[g] = (v8f){bg, bg, bg, bg, bg, bg, bg, bg};
    }

#define WMMA_STEP(APTR, OFS)                                                      \
    do {                                                                          \
        V16U a, b;                                                                \
        a.h[0] = *(const v8h*)((APTR) + (OFS));                                   \
        a.h[1] = *(const v8h*)((APTR) + (OFS) + 16);                              \
        b.h[0] = *(const v8h*)(bp0 + (OFS));                                      \
        b.h[1] = *(const v8h*)(bp0 + (OFS) + 16);                                 \
        acc[0] = __builtin_amdgcn_wmma_f32_16x16x32_f16(false, a.v, false, b.v,   \
                                                        (short)0, acc[0], false, false); \
        b.h[0] = *(const v8h*)(bp1 + (OFS));                                      \
        b.h[1] = *(const v8h*)(bp1 + (OFS) + 16);                                 \
        acc[1] = __builtin_amdgcn_wmma_f32_16x16x32_f16(false, a.v, false, b.v,   \
                                                        (short)0, acc[1], false, false); \
        b.h[0] = *(const v8h*)(bp2 + (OFS));                                      \
        b.h[1] = *(const v8h*)(bp2 + (OFS) + 16);                                 \
        acc[2] = __builtin_amdgcn_wmma_f32_16x16x32_f16(false, a.v, false, b.v,   \
                                                        (short)0, acc[2], false, false); \
        b.h[0] = *(const v8h*)(bp3 + (OFS));                                      \
        b.h[1] = *(const v8h*)(bp3 + (OFS) + 16);                                 \
        acc[3] = __builtin_amdgcn_wmma_f32_16x16x32_f16(false, a.v, false, b.v,   \
                                                        (short)0, acc[3], false, false); \
    } while (0)

    // Recurrent segment: K = 0 .. H-1  (B rows offset 0 in packed weights)
#pragma unroll 4
    for (int kb = 0; kb < H_; kb += 32) {
        WMMA_STEP(ap, kb);
    }
    // Input segment: K = H .. H+KX-1 (A from seq; B rows offset H)
#pragma unroll 4
    for (int kb = 0; kb < KX; kb += 32) {
        V16U a, b;
        a.h[0] = *(const v8h*)(xp + kb);
        a.h[1] = *(const v8h*)(xp + kb + 16);
        b.h[0] = *(const v8h*)(bp0 + H_ + kb);
        b.h[1] = *(const v8h*)(bp0 + H_ + kb + 16);
        acc[0] = __builtin_amdgcn_wmma_f32_16x16x32_f16(false, a.v, false, b.v,
                                                        (short)0, acc[0], false, false);
        b.h[0] = *(const v8h*)(bp1 + H_ + kb);
        b.h[1] = *(const v8h*)(bp1 + H_ + kb + 16);
        acc[1] = __builtin_amdgcn_wmma_f32_16x16x32_f16(false, a.v, false, b.v,
                                                        (short)0, acc[1], false, false);
        b.h[0] = *(const v8h*)(bp2 + H_ + kb);
        b.h[1] = *(const v8h*)(bp2 + H_ + kb + 16);
        acc[2] = __builtin_amdgcn_wmma_f32_16x16x32_f16(false, a.v, false, b.v,
                                                        (short)0, acc[2], false, false);
        b.h[0] = *(const v8h*)(bp3 + H_ + kb);
        b.h[1] = *(const v8h*)(bp3 + H_ + kb + 16);
        acc[3] = __builtin_amdgcn_wmma_f32_16x16x32_f16(false, a.v, false, b.v,
                                                        (short)0, acc[3], false, false);
    }
#undef WMMA_STEP

    // Gate epilogue. C/D layout: VGPR r -> M = m0 + r + (lane<16 ? 0 : 8),
    // N = n0 + (lane & 15).
#pragma unroll
    for (int r = 0; r < 8; ++r) {
        const int m = m0 + r + ((lane < 16) ? 0 : 8);
        const int n = n0 + l15;
        const size_t idx = (size_t)m * H_ + n;
        const float zi = acc[0][r], zf = acc[1][r], zg = acc[2][r], zo = acc[3][r];
        const float cold = c_state[idx];
        const float cnew = sigmoidf_(zf) * cold + sigmoidf_(zi) * tanhf(zg);
        const float hnew = sigmoidf_(zo) * tanhf(cnew);
        c_state[idx] = cnew;
        h_state[idx] = hnew;
        hout[idx]    = (f16t)hnew;
        if (HAS_SEQ_OUT) outseq[((size_t)m * T_ + t) * H_ + n] = (f16t)hnew;
    }
}

// ---------------------------------------------------------------------------
// Heads: mean = h1a_last @ Wm + bm ; value = h1c_last @ Wv + bv ; log_std copy.
// Tiny (2312 outputs x 512-dot) -> scalar kernel.
// ---------------------------------------------------------------------------
__global__ void heads_kernel(const float* __restrict__ h1a,  // as1.h [B,H]
                             const float* __restrict__ h1c,  // cs1.h [B,H]
                             const float* __restrict__ Wv, const float* __restrict__ bv,
                             const float* __restrict__ Wm, const float* __restrict__ bm,
                             const float* __restrict__ lstd,
                             float* __restrict__ out) {
    int idx = blockIdx.x * blockDim.x + threadIdx.x;
    if (idx < B_ * A_) {
        int b = idx >> 3, a = idx & 7;
        float s = 0.f;
        for (int h = 0; h < H_; ++h) s += h1a[(size_t)b * H_ + h] * Wm[(size_t)h * A_ + a];
        out[idx] = s + bm[a];
    } else if (idx < B_ * A_ + A_) {
        out[idx] = lstd[idx - B_ * A_];
    } else if (idx < B_ * A_ + A_ + B_) {
        int b = idx - (B_ * A_ + A_);
        float s = 0.f;
        for (int h = 0; h < H_; ++h) s += h1c[(size_t)b * H_ + h] * Wv[h];
        out[idx] = s + bv[0];
    }
}

// ---------------------------------------------------------------------------
extern "C" void kernel_launch(void* const* d_in, const int* in_sizes, int n_in,
                              void* d_out, int out_size, void* d_ws, size_t ws_size,
                              hipStream_t stream) {
    (void)in_sizes; (void)n_in; (void)out_size; (void)ws_size;

    // ---- inputs (setup_inputs dict order) ----
    const float* x    = (const float*)d_in[0];
    const float* c0c  = (const float*)d_in[1];
    const float* h0c  = (const float*)d_in[2];
    const float* c1c  = (const float*)d_in[3];
    const float* h1c  = (const float*)d_in[4];
    const float* c0a  = (const float*)d_in[5];
    const float* h0a  = (const float*)d_in[6];
    const float* c1a  = (const float*)d_in[7];
    const float* h1a  = (const float*)d_in[8];
    const float* Wx0c = (const float*)d_in[9];
    const float* Wh0c = (const float*)d_in[10];
    const float* b0c  = (const float*)d_in[11];
    const float* Wx1c = (const float*)d_in[12];
    const float* Wh1c = (const float*)d_in[13];
    const float* b1c  = (const float*)d_in[14];
    const float* Wx0a = (const float*)d_in[15];
    const float* Wh0a = (const float*)d_in[16];
    const float* b0a  = (const float*)d_in[17];
    const float* Wx1a = (const float*)d_in[18];
    const float* Wh1a = (const float*)d_in[19];
    const float* b1a  = (const float*)d_in[20];
    const float* Wv   = (const float*)d_in[21];
    const float* bv   = (const float*)d_in[22];
    const float* Wm   = (const float*)d_in[23];
    const float* bm   = (const float*)d_in[24];
    const float* lstd = (const float*)d_in[25];

    float* out = (float*)d_out;

    // ---- output layout (flattened return order) ----
    const size_t BH = (size_t)B_ * H_;            // 131072
    float* o_as0c = out + 2312 + 0 * BH;          // actor L0 c
    float* o_as0h = out + 2312 + 1 * BH;          // actor L0 h
    float* o_cs0c = out + 2312 + 2 * BH;          // critic L0 c
    float* o_cs0h = out + 2312 + 3 * BH;          // critic L0 h
    float* o_as1c = out + 2312 + 4 * BH;          // actor L1 c
    float* o_as1h = out + 2312 + 5 * BH;          // actor L1 h
    float* o_cs1c = out + 2312 + 6 * BH;          // critic L1 c
    float* o_cs1h = out + 2312 + 7 * BH;          // critic L1 h

    // ---- workspace carve-out ----
    char* ws = (char*)d_ws;
    size_t off = 0;
    auto alloc = [&](size_t bytes) -> void* {
        void* p = ws + off;
        off += (bytes + 255) & ~(size_t)255;
        return p;
    };
    const int Kt0 = H_ + D_;   // 768  (layer 0 combined K)
    const int Kt1 = H_ + H_;   // 1024 (layer 1 combined K)

    f16t* x16   = (f16t*)alloc((size_t)B_ * T_ * D_ * sizeof(f16t));
    f16t* Wc0c  = (f16t*)alloc((size_t)4 * H_ * Kt0 * sizeof(f16t));
    f16t* Wc0a  = (f16t*)alloc((size_t)4 * H_ * Kt0 * sizeof(f16t));
    f16t* Wc1c  = (f16t*)alloc((size_t)4 * H_ * Kt1 * sizeof(f16t));
    f16t* Wc1a  = (f16t*)alloc((size_t)4 * H_ * Kt1 * sizeof(f16t));
    f16t* ys0c  = (f16t*)alloc((size_t)B_ * T_ * H_ * sizeof(f16t));
    f16t* ys0a  = (f16t*)alloc((size_t)B_ * T_ * H_ * sizeof(f16t));
    f16t* hb0c[2], *hb0a[2], *hb1c[2], *hb1a[2];
    for (int i = 0; i < 2; ++i) hb0c[i] = (f16t*)alloc(BH * sizeof(f16t));
    for (int i = 0; i < 2; ++i) hb0a[i] = (f16t*)alloc(BH * sizeof(f16t));
    for (int i = 0; i < 2; ++i) hb1c[i] = (f16t*)alloc(BH * sizeof(f16t));
    for (int i = 0; i < 2; ++i) hb1a[i] = (f16t*)alloc(BH * sizeof(f16t));

    // ---- init live c/h state in d_out from inputs ----
    const size_t BHb = BH * sizeof(float);
    hipMemcpyAsync(o_as0c, c0a, BHb, hipMemcpyDeviceToDevice, stream);
    hipMemcpyAsync(o_as0h, h0a, BHb, hipMemcpyDeviceToDevice, stream);
    hipMemcpyAsync(o_cs0c, c0c, BHb, hipMemcpyDeviceToDevice, stream);
    hipMemcpyAsync(o_cs0h, h0c, BHb, hipMemcpyDeviceToDevice, stream);
    hipMemcpyAsync(o_as1c, c1a, BHb, hipMemcpyDeviceToDevice, stream);
    hipMemcpyAsync(o_as1h, h1a, BHb, hipMemcpyDeviceToDevice, stream);
    hipMemcpyAsync(o_cs1c, c1c, BHb, hipMemcpyDeviceToDevice, stream);
    hipMemcpyAsync(o_cs1h, h1c, BHb, hipMemcpyDeviceToDevice, stream);

    // ---- conversions ----
    {
        int n = B_ * T_ * D_;
        cvt_f32_f16_kernel<<<(n + 255) / 256, 256, 0, stream>>>(x, x16, n);
        int m = (int)BH;
        cvt_f32_f16_kernel<<<(m + 255) / 256, 256, 0, stream>>>(h0c, hb0c[0], m);
        cvt_f32_f16_kernel<<<(m + 255) / 256, 256, 0, stream>>>(h0a, hb0a[0], m);
        cvt_f32_f16_kernel<<<(m + 255) / 256, 256, 0, stream>>>(h1c, hb1c[0], m);
        cvt_f32_f16_kernel<<<(m + 255) / 256, 256, 0, stream>>>(h1a, hb1a[0], m);
    }

    // ---- pack + transpose combined weights [Wh;Wx]^T -> f16 ----
    {
        int n0 = 4 * H_ * Kt0, n1 = 4 * H_ * Kt1;
        pack_weights_kernel<<<(n0 + 255) / 256, 256, 0, stream>>>(Wh0c, Wx0c, Wc0c, H_, D_);
        pack_weights_kernel<<<(n0 + 255) / 256, 256, 0, stream>>>(Wh0a, Wx0a, Wc0a, H_, D_);
        pack_weights_kernel<<<(n1 + 255) / 256, 256, 0, stream>>>(Wh1c, Wx1c, Wc1c, H_, H_);
        pack_weights_kernel<<<(n1 + 255) / 256, 256, 0, stream>>>(Wh1a, Wx1a, Wc1a, H_, H_);
    }

    const dim3 sgrid(64, 2), sblk(256);

    // ---- layer 0 scan (critic + actor fused via gridDim.y), stores ys0 f16 ----
    for (int t = 0; t < T_; ++t) {
        int pi = t & 1, po = pi ^ 1;
        lstm_step_kernel<D_, true><<<sgrid, sblk, 0, stream>>>(
            hb0c[pi], hb0a[pi], x16, x16, Wc0c, Wc0a, b0c, b0a,
            o_cs0c, o_as0c, o_cs0h, o_as0h, hb0c[po], hb0a[po], ys0c, ys0a, t);
    }

    // ---- layer 1 scan (critic + actor fused via gridDim.y) ----
    for (int t = 0; t < T_; ++t) {
        int pi = t & 1, po = pi ^ 1;
        lstm_step_kernel<H_, false><<<sgrid, sblk, 0, stream>>>(
            hb1c[pi], hb1a[pi], ys0c, ys0a, Wc1c, Wc1a, b1c, b1a,
            o_cs1c, o_as1c, o_cs1h, o_as1h, hb1c[po], hb1a[po], nullptr, nullptr, t);
    }

    // ---- heads ----
    {
        int n = B_ * A_ + A_ + B_;   // 2312
        heads_kernel<<<(n + 255) / 256, 256, 0, stream>>>(
            o_as1h, o_cs1h, Wv, bv, Wm, bm, lstd, out);
    }
}
